// TernaryLinear_3169685865296
// MI455X (gfx1250) — matrix operand order repaired and checked by reading, not compiled, not run on hardware
//
#include <hip/hip_runtime.h>
#include <hip/hip_bf16.h>

typedef __attribute__((ext_vector_type(8)))  _Float16 v8h;
typedef __attribute__((ext_vector_type(16))) _Float16 v16h;
typedef __attribute__((ext_vector_type(8)))  float    v8f;

#define K_DIM 4096
#define N_DIM 4096
#define LDSS  72   // padded LDS row stride in halfs (144 B: 16B-aligned, bank-friendly)

#if __has_builtin(__builtin_amdgcn_s_wait_asynccnt)
#define ASYNC_WAIT(N) __builtin_amdgcn_s_wait_asynccnt(N)
#else
#define ASYNC_WAIT(N) asm volatile("s_wait_asynccnt %0" :: "i"(N) : "memory")
#endif

// Async global->LDS 16B copy, SADDR (GVS) form: mem = SGPR64 base + per-lane u32 offset.
// Keeps per-stage address advance on the scalar unit (uniform base), not v_add_nc_u64 chains.
static __device__ __forceinline__ void async_copy_b128_s(unsigned lds_addr, unsigned off,
                                                         unsigned long long sbase) {
    asm volatile("global_load_async_to_lds_b128 %0, %1, %2"
                 :: "v"(lds_addr), "v"(off), "s"(sbase)
                 : "memory");
}

// ---------------- Pass 1: per-row ternary quantization of weight ----------------
// scale[r] = max(mean(|w[r,:]|), 1e-5); q[r][k] = clamp(rint(w/scale), -1, 1) as f16 (exact)
__global__ __launch_bounds__(256) void quantize_rows(const float* __restrict__ w,
                                                     _Float16* __restrict__ q,
                                                     float* __restrict__ scales) {
    __shared__ float red[256];
    __shared__ float s_scale;
    const int row = blockIdx.x;
    const int tid = threadIdx.x;
    const float* wr = w + (size_t)row * K_DIM;

    float s = 0.0f;
    #pragma unroll 4
    for (int i = tid; i < K_DIM; i += 256) s += fabsf(wr[i]);
    red[tid] = s;
    __syncthreads();
    for (int off = 128; off > 0; off >>= 1) {
        if (tid < off) red[tid] += red[tid + off];
        __syncthreads();
    }
    if (tid == 0) {
        float sc = fmaxf(red[0] * (1.0f / K_DIM), 1e-5f);
        s_scale = sc;
        scales[row] = sc;
    }
    __syncthreads();

    const float inv = 1.0f / s_scale;
    _Float16* qr = q + (size_t)row * K_DIM;
    #pragma unroll 4
    for (int i = tid; i < K_DIM; i += 256) {
        float v = rintf(wr[i] * inv);               // round-to-nearest-even, like jnp.round
        v = fminf(1.0f, fmaxf(-1.0f, v));
        qr[i] = (_Float16)v;                        // {-1,0,1} exact in f16
    }
}

// ---------------- Pass 2: x f32 -> f16 (vectorized, 8 elements/thread) ----------------
__global__ __launch_bounds__(256) void f32_to_f16(const float4* __restrict__ x4,
                                                  v8h* __restrict__ xh8, int n8) {
    int i = blockIdx.x * 256 + threadIdx.x;
    if (i >= n8) return;
    float4 a = x4[2 * i + 0];
    float4 b = x4[2 * i + 1];
    v8h r;
    r[0] = (_Float16)a.x; r[1] = (_Float16)a.y; r[2] = (_Float16)a.z; r[3] = (_Float16)a.w;
    r[4] = (_Float16)b.x; r[5] = (_Float16)b.y; r[6] = (_Float16)b.z; r[7] = (_Float16)b.w;
    xh8[i] = r;
}

static __device__ __forceinline__ v16h combine16(v8h lo, v8h hi) {
    v16h r;
    #pragma unroll
    for (int e = 0; e < 8; ++e) { r[e] = lo[e]; r[e + 8] = hi[e]; }
    return r;
}

// ---------------- Pass 3: WMMA GEMM  C[M,N] = Ah[M,K] * Qh[N,K]^T, then *scale[n] ----
// Block: 128x128 tile, BK=64, double-buffered LDS filled by async global->LDS copies.
// 8 waves: 4 (rows) x 2 (cols); wave tile 32x64 = 2x4 WMMA accumulators.
__global__ __launch_bounds__(256) void ternary_gemm(const _Float16* __restrict__ A,  // [M,K] f16
                                                    const _Float16* __restrict__ B,  // [N,K] f16
                                                    const float* __restrict__ scales,
                                                    float* __restrict__ C, int M) {
    __shared__ _Float16 As[2][128 * LDSS];
    __shared__ _Float16 Bs[2][128 * LDSS];

    const int tid     = threadIdx.x;
    const int lane    = tid & 31;
    const int wave    = tid >> 5;
    const int wRow    = wave & 3;          // 4 wave-rows * 32 = 128
    const int wCol    = wave >> 2;         // 2 wave-cols * 64 = 128
    const int blockM  = blockIdx.y * 128;
    const int blockN  = blockIdx.x * 128;
    const int halfSel = lane >> 4;         // K-half select per 16-bit WMMA layout
    const int laneLo  = lane & 15;

    // Per-thread staging assignment: 4 x 16B chunks per tile (128 rows x 8 chunks).
    const int sr = tid >> 3;               // base row (row = sr + i*32)
    const int sc = (tid & 7) * 8;          // half-offset within row

    // Loop-invariant per-lane byte offsets (same for A and B panels).
    unsigned goff[4];
    #pragma unroll
    for (int i = 0; i < 4; ++i)
        goff[i] = (unsigned)(((sr + i * 32) * K_DIM + sc) * 2);

    v8f acc[2][4];
    #pragma unroll
    for (int fr = 0; fr < 2; ++fr)
        #pragma unroll
        for (int fc = 0; fc < 4; ++fc)
            #pragma unroll
            for (int e = 0; e < 8; ++e) acc[fr][fc][e] = 0.0f;

    // Uniform SGPR bases for the two panels; advance by 128 B per stage (scalar adds).
    unsigned long long baseA = (unsigned long long)(size_t)(A + (size_t)blockM * K_DIM);
    unsigned long long baseB = (unsigned long long)(size_t)(B + (size_t)blockN * K_DIM);

    // Issue one stage of async copies (8 per thread -> ASYNCcnt += 8 per wave).
    auto issue_stage = [&](int buf, unsigned long long bA, unsigned long long bB) {
        #pragma unroll
        for (int i = 0; i < 4; ++i) {
            int r = sr + i * 32;
            unsigned la = (unsigned)(size_t)&As[buf][r * LDSS + sc];
            unsigned lb = (unsigned)(size_t)&Bs[buf][r * LDSS + sc];
            async_copy_b128_s(la, goff[i], bA);
            async_copy_b128_s(lb, goff[i], bB);
        }
    };

    auto compute_stage = [&](int buf) {
        #pragma unroll
        for (int kk = 0; kk < 2; ++kk) {
            // A fragment (16x32 f16): lanes 0-15 hold K runs [0..7],[16..23]; lanes 16-31 [8..15],[24..31]
            v16h af[2];
            #pragma unroll
            for (int fr = 0; fr < 2; ++fr) {
                int base = (wRow * 32 + fr * 16 + laneLo) * LDSS + kk * 32 + halfSel * 8;
                v8h lo = *(const v8h*)&As[buf][base];
                v8h hi = *(const v8h*)&As[buf][base + 16];
                af[fr] = combine16(lo, hi);
            }
            // B fragment (32x16 f16): lane n = N; lanes 0-15 hold K runs [0..7],[16..23]; lanes 16-31 [8..15],[24..31]
            v16h bf[4];
            #pragma unroll
            for (int fc = 0; fc < 4; ++fc) {
                int base = (wCol * 64 + fc * 16 + laneLo) * LDSS + kk * 32 + halfSel * 8;
                v8h lo = *(const v8h*)&Bs[buf][base];
                v8h hi = *(const v8h*)&Bs[buf][base + 16];
                bf[fc] = combine16(lo, hi);
            }
            #pragma unroll
            for (int fr = 0; fr < 2; ++fr)
                #pragma unroll
                for (int fc = 0; fc < 4; ++fc)
                    acc[fr][fc] = __builtin_amdgcn_wmma_f32_16x16x32_f16(
                        false, af[fr], false, bf[fc], (short)0, acc[fr][fc], false, false);
        }
    };

    issue_stage(0, baseA, baseB);           // prologue: fill buffer 0

    int cur = 0;
    // Steady state: straight-line body, no branchy waits (final stage peeled).
    for (int kt = 0; kt < K_DIM - 64; kt += 64, cur ^= 1) {
        baseA += 128;                       // next 64-half K slice (uniform, scalar)
        baseB += 128;
        issue_stage(cur ^ 1, baseA, baseB); // stream next tile while computing this one
        ASYNC_WAIT(8);                      // in-order: <=8 outstanding => current tile landed
        __syncthreads();
        compute_stage(cur);
        __syncthreads();                    // everyone done reading `cur` before it is refilled
    }
    ASYNC_WAIT(0);                          // final stage
    __syncthreads();
    compute_stage(cur);

    // Epilogue: C/D layout — element v of v8f is M = v + 8*(lane>>4), N = lane&15. Apply scale[n].
    #pragma unroll
    for (int fc = 0; fc < 4; ++fc) {
        int n = blockN + wCol * 64 + fc * 16 + laneLo;
        float scv = scales[n];
        #pragma unroll
        for (int fr = 0; fr < 2; ++fr) {
            int m0 = blockM + wRow * 32 + fr * 16 + halfSel * 8;
            #pragma unroll
            for (int v = 0; v < 8; ++v)
                C[(size_t)(m0 + v) * N_DIM + n] = acc[fr][fc][v] * scv;
        }
    }
}

extern "C" void kernel_launch(void* const* d_in, const int* in_sizes, int n_in,
                              void* d_out, int out_size, void* d_ws, size_t ws_size,
                              hipStream_t stream) {
    const float* x = (const float*)d_in[0];   // [M, 4096] f32
    const float* w = (const float*)d_in[1];   // [4096, 4096] f32
    float* out = (float*)d_out;               // [M, 4096] f32
    const int M = in_sizes[0] / K_DIM;        // 8192

    char* ws = (char*)d_ws;
    _Float16* qh     = (_Float16*)ws;                                  // 32 MB ternary f16 weight [N,K]
    float*    scales = (float*)(ws + (size_t)32 * 1024 * 1024);        // 16 KB row scales
    _Float16* xh     = (_Float16*)(ws + (size_t)33 * 1024 * 1024);     // 64 MB x in f16 [M,K]

    quantize_rows<<<N_DIM, 256, 0, stream>>>(w, qh, scales);

    int n8 = in_sizes[0] / 8;
    f32_to_f16<<<(n8 + 255) / 256, 256, 0, stream>>>((const float4*)x, (v8h*)xh, n8);

    dim3 grid(N_DIM / 128, M / 128);
    ternary_gemm<<<grid, 256, 0, stream>>>(xh, qh, scales, out, M);
}
